// BottleNeck_Long_61314953118038
// MI455X (gfx1250) — matrix operand order repaired and checked by reading, not compiled, not run on hardware
//
#include <hip/hip_runtime.h>
#include <math.h>

// ---- problem constants (from reference) ----
#define EPSV  1e-5f
#define BB    4
#define CIN   64
#define HH    96
#define WWI   96
#define HW    (HH * WWI)        // 9216
#define CR    16                // reduced channels
#define KS    7
#define KKT   49                // K*K
#define NG    4                 // involution groups
#define KKG   (KKT * NG)        // 196
#define COUT  128
#define NPIX  (BB * HW)         // 36864
#define NTILE (NPIX / 16)       // 2304 16-pixel tiles

typedef __attribute__((ext_vector_type(16))) __bf16        v16bf;
typedef __attribute__((ext_vector_type(8)))  float         v8f;
typedef __attribute__((ext_vector_type(4)))  unsigned int  v4u;
typedef __attribute__((ext_vector_type(8)))  int           v8i;
typedef __attribute__((ext_vector_type(4)))  int           v4i;

__device__ __forceinline__ float geluf(float v) {
    return 0.5f * v * (1.0f + erff(v * 0.70710678118654752f));
}

// ---------------------------------------------------------------------------
// Kernel 1: per-pixel involution kernels.
//   t    = relu(BN(w_reduce @ x))            [16 x 16px]  (WMMA, K=64)
//   kern = w_span @ t + b_span               [196 x 16px] (WMMA, K=16 padded)
// One wave per 16-pixel tile; 4 waves per block.
// ---------------------------------------------------------------------------
__global__ __launch_bounds__(128) void k_kern(
    const float* __restrict__ x, const float* __restrict__ w_reduce,
    const float* __restrict__ g_r, const float* __restrict__ b_r,
    const float* __restrict__ m_r, const float* __restrict__ v_r,
    const float* __restrict__ w_span, const float* __restrict__ b_span,
    float* __restrict__ kern_ws)
{
    __shared__ float tsh[4 * 256];             // per-wave 16x16 t tile (f32)
    const int lane = threadIdx.x & 31;
    const int wv   = threadIdx.x >> 5;
    const int half = lane >> 4;
    const int lm   = lane & 15;
    const int tile = blockIdx.x * 4 + wv;
    const int p0   = tile * 16;
    const int b    = p0 / HW;
    const int pix  = (p0 % HW) + lm;           // column n = lm
    const float* xb = x + (size_t)b * CIN * HW;

    // ---- reduce GEMM: fold BN scale into A rows ----
    const float sr_row = g_r[lm] * rsqrtf(v_r[lm] + EPSV);
    v8f acc = {0.f, 0.f, 0.f, 0.f, 0.f, 0.f, 0.f, 0.f};
#pragma unroll
    for (int s = 0; s < 2; ++s) {
        v16bf a, bx;
#pragma unroll
        for (int e = 0; e < 16; ++e) {
            const int ka = s * 32 + half * 16 + e;                          // A layout
            a[e] = (__bf16)(w_reduce[lm * CIN + ka] * sr_row);
            const int kb = s * 32 + half * 16 + e;                          // B layout
            bx[e] = (__bf16)xb[(size_t)kb * HW + pix];
        }
        acc = __builtin_amdgcn_wmma_f32_16x16x32_bf16(false, a, false, bx,
                                                      (short)0, acc, false, false);
    }

    // ---- relu(BN) epilogue, stash t into LDS ----
#pragma unroll
    for (int r = 0; r < 8; ++r) {
        const int   cr  = r + 8 * half;        // D row
        const float s2  = g_r[cr] * rsqrtf(v_r[cr] + EPSV);
        const float off = b_r[cr] - m_r[cr] * s2;
        tsh[wv * 256 + cr * 16 + lm] = fmaxf(acc[r] + off, 0.f);
    }
    __syncthreads();

    // ---- B operand for span GEMM: K=16 real, padded to 32 with zeros.
    // Unconditional ds_load + value mask (no per-lane exec branching).
    const float hmask = (half == 0) ? 1.f : 0.f;
    v16bf bt;
#pragma unroll
    for (int e = 0; e < 16; ++e)
        bt[e] = (__bf16)(tsh[wv * 256 + e * 16 + lm] * hmask);

    float* kw = kern_ws + (size_t)b * KKG * HW;
    for (int mt = 0; mt < 13; ++mt) {          // ceil(196/16) M-tiles
        const int   row   = mt * 16 + lm;
        const int   rowc  = row < KKG ? row : (KKG - 1);   // clamp, mask value
        const float amask = row < KKG ? 1.f : 0.f;
        v16bf a2;
#pragma unroll
        for (int e = 0; e < 16; ++e) {
            float wsv = 0.f;
            if (half == 0) wsv = w_span[rowc * CR + e] * amask;
            a2[e] = (__bf16)wsv;               // K>=16 elements stay zero
        }
        v8f c2 = {0.f, 0.f, 0.f, 0.f, 0.f, 0.f, 0.f, 0.f};
        c2 = __builtin_amdgcn_wmma_f32_16x16x32_bf16(false, a2, false, bt,
                                                     (short)0, c2, false, false);
#pragma unroll
        for (int r = 0; r < 8; ++r) {
            const int o = mt * 16 + r + 8 * half;
            if (o < KKG) kw[(size_t)o * HW + pix] = c2[r] + b_span[o];
        }
    }
}

// ---------------------------------------------------------------------------
// Kernel 2: involution gather (7x7, 4 groups of 16 channels) + BN + GELU.
// One block = 16x16 spatial tile.
//  - x halo tile (22x22 x 16ch) staged in LDS by cooperative vector loads
//  - per-group kern tile (49 x 16 x 16, fully in-bounds, strides 96/9216)
//    DMA'd into LDS by the Tensor Data Mover, overlapped with the halo load
// ---------------------------------------------------------------------------
__global__ __launch_bounds__(256) void k_invol(
    const float* __restrict__ x, const float* __restrict__ kern_ws,
    const float* __restrict__ g_i, const float* __restrict__ b_i,
    const float* __restrict__ m_i, const float* __restrict__ v_i,
    float* __restrict__ x1_ws)
{
    __shared__ float xs[16][22 * 22 + 1];      // 31,040 B (+1: de-conflict banks)
    __shared__ float ksm[KKT * 256];           // 50,176 B kern tile (TDM dest)
    const int tid = threadIdx.x;
    const int bz  = blockIdx.x;                // b*36 + th*6 + tw
    const int b   = bz / 36;
    const int th  = (bz % 36) / 6;
    const int tw  = bz % 6;
    const int h0  = th * 16, w0 = tw * 16;
    const int lh  = tid >> 4, lw = tid & 15;
    const int pix = (h0 + lh) * WWI + (w0 + lw);
    const float* xb = x       + (size_t)b * CIN * HW;
    const float* kb = kern_ws + (size_t)b * KKG * HW;
    float*       ob = x1_ws   + (size_t)b * CIN * HW;

    for (int g = 0; g < NG; ++g) {
        __syncthreads();                       // previous-group consumers done

        // ---- issue TDM: kern tile [49][16][16] f32 -> ksm (wave 0 only) ----
        if ((threadIdx.x >> 5) == 0) {
            const unsigned lds_off =
                (unsigned)(size_t)(void*)&ksm[0];             // addr[31:0] = LDS offset
            const unsigned long long ga = (unsigned long long)(size_t)
                (const void*)(kb + (size_t)(g * KKT) * HW + h0 * WWI + w0);
            v4u g0;
            g0[0] = 1u;                                       // count=1 (valid user D#)
            g0[1] = lds_off;                                  // lds_addr
            g0[2] = (unsigned)(ga & 0xFFFFFFFFu);             // global_addr[31:0]
            g0[3] = (unsigned)((ga >> 32) & 0x1FFFFFFu)       // global_addr[56:32]
                  | (2u << 30);                               // type=2 (image)
            v8i g1;
            g1[0] = (int)(2u << 16);                          // data_size=4B; mask=0
            g1[1] = (int)(16u << 16);                         // tensor_dim0[15:0]=16
            g1[2] = (int)(16u << 16);                         // dim0 hi=0 | tensor_dim1 lo=16
            g1[3] = (int)(16u << 16);                         // dim1 hi=0 | tile_dim0=16
            g1[4] = (int)(16u | (49u << 16));                 // tile_dim1=16, tile_dim2=49
            g1[5] = (int)96;                                  // tensor_dim0_stride lo (=W)
            g1[6] = (int)(9216u << 16);                       // stride0 hi=0 | stride1 lo=HW
            g1[7] = 0;                                        // stride1 hi
            v4i g2 = {KKT, 0, 0, 0};                          // tensor_dim2=49
            v4i g3 = {0, 0, 0, 0};
            v8i g4 = {0, 0, 0, 0, 0, 0, 0, 0};                // unused extra group
            __builtin_amdgcn_tensor_load_to_lds(g0, g1, g2, g3, g4, 0);
        }

        // ---- cooperative halo load into xs (overlaps the TDM) ----
        for (int idx = tid; idx < 16 * 484; idx += 256) {
            const int ch  = idx / 484, pos = idx % 484;
            const int hh  = h0 + pos / 22 - 3;
            const int ww  = w0 + pos % 22 - 3;
            float v = 0.f;
            if (hh >= 0 && hh < HH && ww >= 0 && ww < WWI)
                v = xb[(size_t)(g * 16 + ch) * HW + hh * WWI + ww];
            xs[ch][pos] = v;
        }
        if (g + 1 < NG)                                       // warm L2 for next group
            __builtin_prefetch(xb + (size_t)((g + 1) * 16) * HW + pix, 0, 1);

        if ((threadIdx.x >> 5) == 0)
            __builtin_amdgcn_s_wait_tensorcnt(0);             // TDM data landed
        __syncthreads();

        // ---- 49-tap involution for 16 channels, all operands in LDS ----
        float acc[16];
#pragma unroll
        for (int ch = 0; ch < 16; ++ch) acc[ch] = 0.f;
        for (int kk = 0; kk < KKT; ++kk) {
            const float kv  = ksm[kk * 256 + tid];            // [kk][lh][lw]
            const int   pos = (lh + kk / KS) * 22 + (lw + kk % KS);
#pragma unroll
            for (int ch = 0; ch < 16; ++ch)
                acc[ch] = fmaf(kv, xs[ch][pos], acc[ch]);
        }
#pragma unroll
        for (int ch = 0; ch < 16; ++ch) {
            const int   c  = g * 16 + ch;
            const float si = g_i[c] * rsqrtf(v_i[c] + EPSV);
            const float oi = b_i[c] - m_i[c] * si;
            ob[(size_t)c * HW + pix] = geluf(acc[ch] * si + oi);
        }
    }
}

// ---------------------------------------------------------------------------
// Kernel 3: out = gelu(BN(w_conv@x1) + BN(w_map@x + b_map)).
// BN scales folded into bf16 weights -> single f32 WMMA accumulator chain.
// 8 waves/block, wave mt owns out-channels [mt*16, mt*16+16) for a 16px tile.
// ---------------------------------------------------------------------------
__global__ __launch_bounds__(256) void k_out(
    const float* __restrict__ x, const float* __restrict__ x1_ws,
    const float* __restrict__ w_conv,
    const float* __restrict__ g_c, const float* __restrict__ b_c,
    const float* __restrict__ m_c, const float* __restrict__ v_c,
    const float* __restrict__ w_map, const float* __restrict__ b_map,
    const float* __restrict__ g_m, const float* __restrict__ b_m,
    const float* __restrict__ m_m, const float* __restrict__ v_m,
    float* __restrict__ out)
{
    const int lane = threadIdx.x & 31;
    const int mt   = threadIdx.x >> 5;         // 0..7 -> M-tile
    const int half = lane >> 4;
    const int lm   = lane & 15;
    const int p0   = blockIdx.x * 16;
    const int b    = p0 / HW;
    const int pix  = (p0 % HW) + lm;
    const float* xb  = x     + (size_t)b * CIN * HW;
    const float* x1b = x1_ws + (size_t)b * CIN * HW;

    const int   row = mt * 16 + lm;            // A row this lane supplies
    const float sc  = g_c[row] * rsqrtf(v_c[row] + EPSV);
    const float smv = g_m[row] * rsqrtf(v_m[row] + EPSV);

    v8f acc = {0.f, 0.f, 0.f, 0.f, 0.f, 0.f, 0.f, 0.f};
#pragma unroll
    for (int s = 0; s < 2; ++s) {
        v16bf ac, am, b1, b0;
#pragma unroll
        for (int e = 0; e < 16; ++e) {
            const int ka = s * 32 + half * 16 + e;
            ac[e] = (__bf16)(w_conv[row * CIN + ka] * sc);
            am[e] = (__bf16)(w_map [row * CIN + ka] * smv);
            const int kb = s * 32 + half * 16 + e;
            b1[e] = (__bf16)x1b[(size_t)kb * HW + pix];
            b0[e] = (__bf16)xb [(size_t)kb * HW + pix];
        }
        acc = __builtin_amdgcn_wmma_f32_16x16x32_bf16(false, ac, false, b1,
                                                      (short)0, acc, false, false);
        acc = __builtin_amdgcn_wmma_f32_16x16x32_bf16(false, am, false, b0,
                                                      (short)0, acc, false, false);
    }
#pragma unroll
    for (int r = 0; r < 8; ++r) {
        const int   oc = mt * 16 + r + 8 * half;
        const float s1 = g_c[oc] * rsqrtf(v_c[oc] + EPSV);
        const float s2 = g_m[oc] * rsqrtf(v_m[oc] + EPSV);
        const float bias = (b_c[oc] - m_c[oc] * s1)
                         + (b_m[oc] - m_m[oc] * s2) + b_map[oc] * s2;
        out[(size_t)b * COUT * HW + (size_t)oc * HW + pix] = geluf(acc[r] + bias);
    }
}

// ---------------------------------------------------------------------------
extern "C" void kernel_launch(void* const* d_in, const int* in_sizes, int n_in,
                              void* d_out, int out_size, void* d_ws, size_t ws_size,
                              hipStream_t stream) {
    (void)in_sizes; (void)n_in; (void)out_size; (void)ws_size;
    const float* x        = (const float*)d_in[0];
    const float* w_reduce = (const float*)d_in[1];
    const float* g_r = (const float*)d_in[2];
    const float* b_r = (const float*)d_in[3];
    const float* m_r = (const float*)d_in[4];
    const float* v_r = (const float*)d_in[5];
    const float* w_span = (const float*)d_in[6];
    const float* b_span = (const float*)d_in[7];
    const float* g_i = (const float*)d_in[8];
    const float* b_i = (const float*)d_in[9];
    const float* m_i = (const float*)d_in[10];
    const float* v_i = (const float*)d_in[11];
    const float* w_conv = (const float*)d_in[12];
    const float* g_c = (const float*)d_in[13];
    const float* b_c = (const float*)d_in[14];
    const float* m_c = (const float*)d_in[15];
    const float* v_c = (const float*)d_in[16];
    const float* w_map = (const float*)d_in[17];
    const float* b_map = (const float*)d_in[18];
    const float* g_m = (const float*)d_in[19];
    const float* b_m = (const float*)d_in[20];
    const float* m_m = (const float*)d_in[21];
    const float* v_m = (const float*)d_in[22];
    float* out = (float*)d_out;

    // workspace: kern [B,196,H,W] then x1 [B,64,H,W] (both f32) ~36.6 MB
    float* kern_ws = (float*)d_ws;
    float* x1_ws   = kern_ws + (size_t)BB * KKG * HW;

    k_kern<<<NTILE / 4, 128, 0, stream>>>(x, w_reduce, g_r, b_r, m_r, v_r,
                                          w_span, b_span, kern_ws);
    k_invol<<<BB * 36, 256, 0, stream>>>(x, kern_ws, g_i, b_i, m_i, v_i, x1_ws);
    k_out<<<NTILE, 256, 0, stream>>>(x, x1_ws, w_conv, g_c, b_c, m_c, v_c,
                                     w_map, b_map, g_m, b_m, m_m, v_m, out);
}
